// BaselineOverlapMoE_30777735643616
// MI455X (gfx1250) — matrix operand order repaired and checked by reading, not compiled, not run on hardware
//
#include <hip/hip_runtime.h>

typedef __attribute__((ext_vector_type(16))) __bf16 v16bf;
typedef __attribute__((ext_vector_type(8)))  float  v8f;

#define T_TOK 32768
#define H_DIM 1024
#define N_EXP 4
#define BM 128
#define BN 128
#define BK 64
#define LDS_S 72   // ushort stride per tile row: 64 data + 8 pad, 144B, 16B-aligned

#if defined(__has_builtin)
# if __has_builtin(__builtin_amdgcn_global_load_async_to_lds_b128) && \
     __has_builtin(__builtin_amdgcn_s_wait_asynccnt)
#  define USE_ASYNC 1
# endif
#endif
#ifndef USE_ASYNC
# define USE_ASYNC 0
#endif

struct U4x2 { uint4 a, b; };

__device__ __forceinline__ unsigned short f2bf(float f) {
    unsigned int u = __builtin_bit_cast(unsigned int, f);
    unsigned int r = u + 0x7fffu + ((u >> 16) & 1u);   // RNE
    return (unsigned short)(r >> 16);
}
__device__ __forceinline__ unsigned int pack2(float a, float b) {
    return (unsigned int)f2bf(a) | ((unsigned int)f2bf(b) << 16);
}

#if USE_ASYNC
typedef int v4i __attribute__((vector_size(4 * sizeof(int))));
typedef __attribute__((address_space(1))) v4i gv4i;   // global
typedef __attribute__((address_space(3))) v4i lv4i;   // LDS
// copy 16 ushorts (32B) global -> LDS via two async b128 transfers
__device__ __forceinline__ void async_cp64B(unsigned short* ldst, const unsigned short* gsrc) {
    __builtin_amdgcn_global_load_async_to_lds_b128((gv4i*)gsrc,       (lv4i*)ldst,       0, 0);
    __builtin_amdgcn_global_load_async_to_lds_b128((gv4i*)(gsrc + 8), (lv4i*)(ldst + 8), 0, 0);
}
#endif

// -------- init / convert / gate / prefix / scatter --------

__global__ void moe_init(int* counts) {
    if (threadIdx.x < N_EXP) counts[threadIdx.x] = 0;
}

__global__ void moe_cvt_bf16(const float* __restrict__ src,
                             unsigned short* __restrict__ dst, int n) {
    int i = blockIdx.x * blockDim.x + threadIdx.x;
    if (i < n) dst[i] = f2bf(src[i]);
}

// one wave (32 lanes) per token: 4 dots of length 1024, shfl reduce, argmax.
// Also emits the bf16 copy of the token row (tokens are already streaming here).
__global__ void moe_gate(const float* __restrict__ tokens,
                         const float* __restrict__ gate_w,
                         int* __restrict__ top1, int* __restrict__ counts,
                         unsigned short* __restrict__ tokb) {
    int lane = threadIdx.x & 31;
    int wave = threadIdx.x >> 5;
    int t = blockIdx.x * 8 + wave;
    const float4* tk4 = (const float4*)(tokens + (size_t)t * H_DIM);
    const float4* gw4 = (const float4*)gate_w;    // [E][H/4]
    uint2* tb2 = (uint2*)(tokb + (size_t)t * H_DIM);
    float acc0 = 0.f, acc1 = 0.f, acc2 = 0.f, acc3 = 0.f;
#pragma unroll
    for (int i = 0; i < H_DIM / 128; ++i) {
        int idx = lane + i * 32;
        float4 x = tk4[idx];
        uint2 p; p.x = pack2(x.x, x.y); p.y = pack2(x.z, x.w);
        tb2[idx] = p;
        float4 g0 = gw4[0 * 256 + idx];
        float4 g1 = gw4[1 * 256 + idx];
        float4 g2 = gw4[2 * 256 + idx];
        float4 g3 = gw4[3 * 256 + idx];
        acc0 += x.x * g0.x + x.y * g0.y + x.z * g0.z + x.w * g0.w;
        acc1 += x.x * g1.x + x.y * g1.y + x.z * g1.z + x.w * g1.w;
        acc2 += x.x * g2.x + x.y * g2.y + x.z * g2.z + x.w * g2.w;
        acc3 += x.x * g3.x + x.y * g3.y + x.z * g3.z + x.w * g3.w;
    }
#pragma unroll
    for (int off = 16; off > 0; off >>= 1) {
        acc0 += __shfl_xor(acc0, off, 32);
        acc1 += __shfl_xor(acc1, off, 32);
        acc2 += __shfl_xor(acc2, off, 32);
        acc3 += __shfl_xor(acc3, off, 32);
    }
    if (lane == 0) {
        float best = acc0; int be = 0;
        if (acc1 > best) { best = acc1; be = 1; }
        if (acc2 > best) { best = acc2; be = 2; }
        if (acc3 > best) { best = acc3; be = 3; }
        top1[t] = be;
        atomicAdd(&counts[be], 1);
    }
}

__global__ void moe_prefix(const int* __restrict__ counts, int* __restrict__ cursors) {
    if (threadIdx.x == 0) {
        int s = 0;
        for (int e = 0; e < N_EXP; ++e) { cursors[e] = s; s += counts[e]; }
    }
}

__global__ void moe_scatter(const int* __restrict__ top1, int* __restrict__ cursors,
                            int* __restrict__ perm) {
    int t = blockIdx.x * blockDim.x + threadIdx.x;
    int e = top1[t];
    int pos = atomicAdd(&cursors[e], 1);
    perm[pos] = t;
}

// -------- WMMA fragment loaders (ISA 7.12.2 layouts) --------

__device__ __forceinline__ v16bf ld_fragA(const unsigned short* lA, int rowBase, int koff, int lane) {
    int m = lane & 15;
    int k0 = (lane < 16) ? 0 : 8;                 // A: {0..7,16..23} / {8..15,24..31}
    const unsigned short* p = lA + (rowBase + m) * LDS_S + koff + k0;
    U4x2 u; u.a = *(const uint4*)p; u.b = *(const uint4*)(p + 16);
    return __builtin_bit_cast(v16bf, u);
}
__device__ __forceinline__ v16bf ld_fragB(const unsigned short* lB, int colBase, int koff, int lane) {
    int n = lane & 15;
    int k0 = (lane < 16) ? 0 : 8;                 // B: {0..7,16..23} / {8..15,24..31}
    const unsigned short* p = lB + (colBase + n) * LDS_S + koff + k0;
    U4x2 u; u.a = *(const uint4*)p; u.b = *(const uint4*)(p + 16);
    return __builtin_bit_cast(v16bf, u);
}

// one wave's 32x64 MMA step on a 32-wide K slice of the current LDS buffers
__device__ __forceinline__ void mma_step(const unsigned short* lA, const unsigned short* lB,
                                         int wRow, int wCol, int koff, int lane, v8f acc[2][4]) {
    v16bf a0 = ld_fragA(lA, wRow,      koff, lane);
    v16bf a1 = ld_fragA(lA, wRow + 16, koff, lane);
    v16bf b0 = ld_fragB(lB, wCol,      koff, lane);
    v16bf b1 = ld_fragB(lB, wCol + 16, koff, lane);
    v16bf b2 = ld_fragB(lB, wCol + 32, koff, lane);
    v16bf b3 = ld_fragB(lB, wCol + 48, koff, lane);
    acc[0][0] = __builtin_amdgcn_wmma_f32_16x16x32_bf16(false, a0, false, b0, (short)0, acc[0][0], false, false);
    acc[0][1] = __builtin_amdgcn_wmma_f32_16x16x32_bf16(false, a0, false, b1, (short)0, acc[0][1], false, false);
    acc[0][2] = __builtin_amdgcn_wmma_f32_16x16x32_bf16(false, a0, false, b2, (short)0, acc[0][2], false, false);
    acc[0][3] = __builtin_amdgcn_wmma_f32_16x16x32_bf16(false, a0, false, b3, (short)0, acc[0][3], false, false);
    acc[1][0] = __builtin_amdgcn_wmma_f32_16x16x32_bf16(false, a1, false, b0, (short)0, acc[1][0], false, false);
    acc[1][1] = __builtin_amdgcn_wmma_f32_16x16x32_bf16(false, a1, false, b1, (short)0, acc[1][1], false, false);
    acc[1][2] = __builtin_amdgcn_wmma_f32_16x16x32_bf16(false, a1, false, b2, (short)0, acc[1][2], false, false);
    acc[1][3] = __builtin_amdgcn_wmma_f32_16x16x32_bf16(false, a1, false, b3, (short)0, acc[1][3], false, false);
}

// -------- expert GEMM: y = gelu(gather(tokb) @ W_e^T + b_e), bf16 out --------

__global__ void __launch_bounds__(256)
moe_expert_gemm(const unsigned short* __restrict__ tokb,  // [T][H] bf16
                const unsigned short* __restrict__ ewb,   // [E][H][H] bf16
                const float* __restrict__ eb,             // [E][H]
                const int* __restrict__ counts,
                const int* __restrict__ perm,
                unsigned short* __restrict__ mid) {
    __shared__ unsigned short lA[2][BM * LDS_S];
    __shared__ unsigned short lB[2][BN * LDS_S];
    __shared__ int sTok[BM];

    // map blockIdx.y -> (expert, tile within expert group)
    int bid = blockIdx.y;
    int e = -1, tile = 0, cnt = 0, start = 0, s = 0;
#pragma unroll
    for (int i = 0; i < N_EXP; ++i) {
        int c = counts[i];
        int tl = (c + BM - 1) >> 7;
        if (e < 0) {
            if (bid < tl) { e = i; tile = bid; cnt = c; start = s; }
            else bid -= tl;
        }
        s += c;
    }
    if (e < 0) return;

    int tid  = threadIdx.x;
    int lane = tid & 31;
    int wave = tid >> 5;
    int wRow = (wave & 3) * 32;       // 4 waves down
    int wCol = (wave >> 2) * 64;      // 2 waves across
    int cb   = blockIdx.x * BN;       // output-column tile base

    if (tid < BM) {
        int gl = tile * BM + tid;
        int idx = start + (gl < cnt ? gl : cnt - 1);   // clamp padded rows
        sTok[tid] = perm[idx];
    }
    __syncthreads();

    v8f acc[2][4];
#pragma unroll
    for (int i = 0; i < 2; ++i)
#pragma unroll
        for (int j = 0; j < 4; ++j) acc[i][j] = (v8f)0.f;

    int r = tid >> 1, half = tid & 1;   // 2 threads per row, 32B each
    const unsigned short* aSrc = tokb + (size_t)sTok[r] * H_DIM + half * 32;
    const unsigned short* bSrc = ewb + ((size_t)e * H_DIM + cb + r) * H_DIM + half * 32;
    unsigned short* dA0 = &lA[0][r * LDS_S + half * 32];
    unsigned short* dA1 = &lA[1][r * LDS_S + half * 32];
    unsigned short* dB0 = &lB[0][r * LDS_S + half * 32];
    unsigned short* dB1 = &lB[1][r * LDS_S + half * 32];

#if USE_ASYNC
    async_cp64B(dA0, aSrc);
    async_cp64B(dB0, bSrc);
#else
    uint4 ra0, ra1, rb0, rb1;
    { const uint4* p = (const uint4*)aSrc; ra0 = p[0]; ra1 = p[1]; }
    { const uint4* p = (const uint4*)bSrc; rb0 = p[0]; rb1 = p[1]; }
#endif

    int buf = 0;
    for (int k0 = 0; k0 < H_DIM; k0 += BK) {
#if !USE_ASYNC
        { uint4* d = (uint4*)(buf ? dA1 : dA0); d[0] = ra0; d[1] = ra1; }
        { uint4* d = (uint4*)(buf ? dB1 : dB0); d[0] = rb0; d[1] = rb1; }
#endif
        if (k0 + BK < H_DIM) {                    // prefetch next stage
#if USE_ASYNC
            async_cp64B(buf ? dA0 : dA1, aSrc + k0 + BK);
            async_cp64B(buf ? dB0 : dB1, bSrc + k0 + BK);
            __builtin_amdgcn_s_wait_asynccnt(4);  // older stage (4 transfers) complete
#else
            const uint4* p = (const uint4*)(aSrc + k0 + BK);
            const uint4* q = (const uint4*)(bSrc + k0 + BK);
            ra0 = p[0]; ra1 = p[1]; rb0 = q[0]; rb1 = q[1];
#endif
        } else {
#if USE_ASYNC
            __builtin_amdgcn_s_wait_asynccnt(0);
#endif
        }
        __syncthreads();
        mma_step(&lA[buf][0], &lB[buf][0], wRow, wCol, 0,  lane, acc);
        mma_step(&lA[buf][0], &lB[buf][0], wRow, wCol, 32, lane, acc);
        buf ^= 1;
    }

    // epilogue: bias + exact gelu, predicated scatter store (bf16)
    int hiAdd = (lane < 16) ? 0 : 8;
#pragma unroll
    for (int rf = 0; rf < 2; ++rf)
#pragma unroll
        for (int cf = 0; cf < 4; ++cf) {
            int col = cb + wCol + cf * 16 + (lane & 15);
            float bias = eb[e * H_DIM + col];
#pragma unroll
            for (int v = 0; v < 8; ++v) {
                int rl = wRow + rf * 16 + v + hiAdd;
                int gl = tile * BM + rl;
                if (gl < cnt) {
                    int tok = sTok[rl];
                    float x = acc[rf][cf][v] + bias;
                    float g = 0.5f * x * (1.0f + erff(x * 0.70710678118654752f));
                    mid[(size_t)tok * H_DIM + col] = f2bf(g);
                }
            }
        }
}

// -------- combine GEMM: out = mid @ combine_w^T, fp32 out --------

__global__ void __launch_bounds__(256)
moe_combine_gemm(const unsigned short* __restrict__ mid,
                 const unsigned short* __restrict__ cwb,   // [H][H] bf16
                 float* __restrict__ out) {
    __shared__ unsigned short lA[2][BM * LDS_S];
    __shared__ unsigned short lB[2][BN * LDS_S];

    int tid  = threadIdx.x;
    int lane = tid & 31;
    int wave = tid >> 5;
    int wRow = (wave & 3) * 32;
    int wCol = (wave >> 2) * 64;
    int row0 = blockIdx.y * BM;
    int cb   = blockIdx.x * BN;

    v8f acc[2][4];
#pragma unroll
    for (int i = 0; i < 2; ++i)
#pragma unroll
        for (int j = 0; j < 4; ++j) acc[i][j] = (v8f)0.f;

    int r = tid >> 1, half = tid & 1;
    const unsigned short* aSrc = mid + (size_t)(row0 + r) * H_DIM + half * 32;
    const unsigned short* bSrc = cwb + (size_t)(cb + r) * H_DIM + half * 32;
    unsigned short* dA0 = &lA[0][r * LDS_S + half * 32];
    unsigned short* dA1 = &lA[1][r * LDS_S + half * 32];
    unsigned short* dB0 = &lB[0][r * LDS_S + half * 32];
    unsigned short* dB1 = &lB[1][r * LDS_S + half * 32];

#if USE_ASYNC
    async_cp64B(dA0, aSrc);
    async_cp64B(dB0, bSrc);
#else
    uint4 ra0, ra1, rb0, rb1;
    { const uint4* p = (const uint4*)aSrc; ra0 = p[0]; ra1 = p[1]; }
    { const uint4* p = (const uint4*)bSrc; rb0 = p[0]; rb1 = p[1]; }
#endif

    int buf = 0;
    for (int k0 = 0; k0 < H_DIM; k0 += BK) {
#if !USE_ASYNC
        { uint4* d = (uint4*)(buf ? dA1 : dA0); d[0] = ra0; d[1] = ra1; }
        { uint4* d = (uint4*)(buf ? dB1 : dB0); d[0] = rb0; d[1] = rb1; }
#endif
        if (k0 + BK < H_DIM) {
#if USE_ASYNC
            async_cp64B(buf ? dA0 : dA1, aSrc + k0 + BK);
            async_cp64B(buf ? dB0 : dB1, bSrc + k0 + BK);
            __builtin_amdgcn_s_wait_asynccnt(4);  // older stage (4 transfers) complete
#else
            const uint4* p = (const uint4*)(aSrc + k0 + BK);
            const uint4* q = (const uint4*)(bSrc + k0 + BK);
            ra0 = p[0]; ra1 = p[1]; rb0 = q[0]; rb1 = q[1];
#endif
        } else {
#if USE_ASYNC
            __builtin_amdgcn_s_wait_asynccnt(0);
#endif
        }
        __syncthreads();
        mma_step(&lA[buf][0], &lB[buf][0], wRow, wCol, 0,  lane, acc);
        mma_step(&lA[buf][0], &lB[buf][0], wRow, wCol, 32, lane, acc);
        buf ^= 1;
    }

    int hiAdd = (lane < 16) ? 0 : 8;
#pragma unroll
    for (int rf = 0; rf < 2; ++rf)
#pragma unroll
        for (int cf = 0; cf < 4; ++cf) {
            int col = cb + wCol + cf * 16 + (lane & 15);
#pragma unroll
            for (int v = 0; v < 8; ++v) {
                int rl = wRow + rf * 16 + v + hiAdd;
                out[(size_t)(row0 + rl) * H_DIM + col] = acc[rf][cf][v];
            }
        }
}

// -------- launcher --------

extern "C" void kernel_launch(void* const* d_in, const int* in_sizes, int n_in,
                              void* d_out, int out_size, void* d_ws, size_t ws_size,
                              hipStream_t stream) {
    const float* tokens    = (const float*)d_in[0];   // [T,H]
    const float* gate_w    = (const float*)d_in[1];   // [E,H]
    const float* expert_w  = (const float*)d_in[2];   // [E,H,H]
    const float* expert_b  = (const float*)d_in[3];   // [E,H]
    const float* combine_w = (const float*)d_in[4];   // [H,H]
    float* out = (float*)d_out;

    char* ws = (char*)d_ws;
    int* counts  = (int*)ws;                                    // 4 ints
    int* cursors = (int*)(ws + 16);                             // 4 ints
    int* top1    = (int*)(ws + 64);                             // T ints
    int* perm    = (int*)(ws + 64 + 4 * (size_t)T_TOK);         // T ints
    unsigned short* tokb = (unsigned short*)(ws + 64 + 8 * (size_t)T_TOK);  // [T][H] bf16
    unsigned short* ewb  = tokb + (size_t)T_TOK * H_DIM;
    unsigned short* cwb  = ewb + (size_t)N_EXP * H_DIM * H_DIM;
    unsigned short* mid  = cwb + (size_t)H_DIM * H_DIM;

    moe_init<<<1, 32, 0, stream>>>(counts);
    moe_cvt_bf16<<<(N_EXP * H_DIM * H_DIM) / 256, 256, 0, stream>>>(expert_w, ewb, N_EXP * H_DIM * H_DIM);
    moe_cvt_bf16<<<(H_DIM * H_DIM) / 256, 256, 0, stream>>>(combine_w, cwb, H_DIM * H_DIM);
    moe_gate<<<T_TOK / 8, 256, 0, stream>>>(tokens, gate_w, top1, counts, tokb);
    moe_prefix<<<1, 1, 0, stream>>>(counts, cursors);
    moe_scatter<<<T_TOK / 256, 256, 0, stream>>>(top1, cursors, perm);
    moe_expert_gemm<<<dim3(H_DIM / BN, T_TOK / BM + N_EXP), 256, 0, stream>>>(
        tokb, ewb, expert_b, counts, perm, mid);
    moe_combine_gemm<<<dim3(H_DIM / BN, T_TOK / BM), 256, 0, stream>>>(mid, cwb, out);
}
